// Mesh1_80985903334295
// MI455X (gfx1250) — matrix operand order repaired and checked by reading, not compiled, not run on hardware
//
#include <hip/hip_runtime.h>
#include <hip/hip_bf16.h>

typedef __attribute__((ext_vector_type(16))) _Float16 v16h;
typedef __attribute__((ext_vector_type(8)))  _Float16 v8h;
typedef __attribute__((ext_vector_type(8)))  float    v8f;

// ---- geometry ----
#define ROWS     64            // nodes per block
#define K1_REAL  195           // concat(spatial 64, structural 131); k=195 = bias column
#define K1_PAD   224           // 7 chunks of 32
#define KC1      7
#define SA1      232           // padded LDS row stride (halves) -> bank spread
#define K2_REAL  131           // k=131 = bias column
#define K2_PAD   160           // 5 chunks of 32
#define KC2      5
#define SV2      168           // padded LDS row stride (halves)
#define OC       256           // output channels
#define NT       16            // 16 column tiles of 16
#define F1       (NT * KC1)    // 112 Wc fragments
#define F2       (NT * KC2)    // 80  Wa fragments

// ---------------------------------------------------------------------------
// Pack Wc / Wa (fp32, [out_c, in_c]) + bias into f16 B-fragments in d_ws.
// Fragment f = ct*KC + kc is 32 lanes x 16 halves, contiguous (1 KB).
// Lane l: n = l%16 (out channel within tile), khalf = l/16.
// Element e: k = kc*32 + (e<8 ? khalf*8 + e : 16 + khalf*8 + e-8)  (ISA 16-bit B 32x16 layout).
// Bias is folded in at k == K_REAL (activations carry 1.0 there).
// ---------------------------------------------------------------------------
__global__ __launch_bounds__(256)
void mesh_pack_weights(const float* __restrict__ Wc, const float* __restrict__ bc,
                       const float* __restrict__ Wa, const float* __restrict__ ba,
                       _Float16* __restrict__ wp) {
    int idx = blockIdx.x * 256 + threadIdx.x;
    int total = (F1 + F2) * 512;
    if (idx >= total) return;
    if (idx < F1 * 512) {
        int f = idx >> 9, rem = idx & 511;
        int lane = rem >> 4, e = rem & 15;
        int ct = f / KC1, kc = f % KC1;
        int n = lane & 15, kh = lane >> 4;
        int k = kc * 32 + (e < 8 ? kh * 8 + e : 16 + kh * 8 + (e - 8));
        int o = ct * 16 + n;
        float v = (k < K1_REAL) ? Wc[o * K1_REAL + k]
                : (k == K1_REAL) ? bc[o] : 0.0f;
        wp[idx] = (_Float16)v;
    } else {
        int j = idx - F1 * 512;
        int f = j >> 9, rem = j & 511;
        int lane = rem >> 4, e = rem & 15;
        int ct = f / KC2, kc = f % KC2;
        int n = lane & 15, kh = lane >> 4;
        int k = kc * 32 + (e < 8 ? kh * 8 + e : 16 + kh * 8 + (e - 8));
        int o = ct * 16 + n;
        float v = (k < K2_REAL) ? Wa[o * K2_REAL + k]
                : (k == K2_REAL) ? ba[o] : 0.0f;
        wp[F1 * 512 + j] = (_Float16)v;
    }
}

// ---------------------------------------------------------------------------
// One wave: 32 rows x 128 cols (2 row tiles x 8 column tiles).
// Each B fragment feeds two independent WMMAs (no RAW chain), B traffic /2.
// Zero-init accumulators (bias folded into K); +1 B-fragment prefetch.
// ---------------------------------------------------------------------------
template<int KC, bool FULL>
__device__ __forceinline__
void wave_gemm(const _Float16* __restrict__ row0,   // LDS, row tile 0 (16 rows)
               const _Float16* __restrict__ row1,   // LDS, row tile 1 (16 rows)
               int rowStride,                       // halves
               const _Float16* __restrict__ wp,     // packed weight fragments
               int ct0,                             // first of 8 column tiles
               float* __restrict__ out,
               long nb0, long nb1, long Ntot, int lane) {
    const int n    = lane & 15;
    const int half = lane >> 4;
    const _Float16* ar0 = row0 + (size_t)n * rowStride;
    const _Float16* ar1 = row1 + (size_t)n * rowStride;

    // A fragments for both row tiles stay resident across all column tiles.
    v16h a0[KC], a1[KC];
#pragma unroll
    for (int kc = 0; kc < KC; ++kc) {
        const v8h l0 = *(const v8h*)(ar0 + kc * 32 + half * 8);
        const v8h h0 = *(const v8h*)(ar0 + kc * 32 + 16 + half * 8);
        const v8h l1 = *(const v8h*)(ar1 + kc * 32 + half * 8);
        const v8h h1 = *(const v8h*)(ar1 + kc * 32 + 16 + half * 8);
#pragma unroll
        for (int e = 0; e < 8; ++e) {
            a0[kc][e] = l0[e]; a0[kc][e + 8] = h0[e];
            a1[kc][e] = l1[e]; a1[kc][e + 8] = h1[e];
        }
    }

    // B fragments linear in t (within this wave's 8 ct range): prefetch t+1.
    const _Float16* bptr = wp + ((size_t)ct0 * KC) * 512 + (size_t)lane * 16;
    v16h b = *(const v16h*)bptr;
    int t = 0;

    for (int c = 0; c < 8; ++c) {
        v8f acc0 = {};   // bias folded into K -> inline-0 C operand
        v8f acc1 = {};
#pragma unroll
        for (int kc = 0; kc < KC; ++kc) {
            const int tn = (t + 1 < 8 * KC) ? (t + 1) : t;
            const v16h bn = *(const v16h*)(bptr + (size_t)tn * 512);
            acc0 = __builtin_amdgcn_wmma_f32_16x16x32_f16(
                       false, a0[kc], false, b, (short)0, acc0, false, false);
            acc1 = __builtin_amdgcn_wmma_f32_16x16x32_f16(
                       false, a1[kc], false, b, (short)0, acc1, false, false);
            b = bn;
            ++t;
        }

        // D layout: m = v + 8*half, col = (ct0+c)*16 + n
        const int col = (ct0 + c) * 16 + n;
        float* op0 = out + (nb0 + (long)half * 8) * OC + col;
        float* op1 = out + (nb1 + (long)half * 8) * OC + col;
        if (FULL) {
#pragma unroll
            for (int v = 0; v < 8; ++v) {
                __builtin_nontemporal_store(acc0[v], op0 + (size_t)v * OC);
                __builtin_nontemporal_store(acc1[v], op1 + (size_t)v * OC);
            }
        } else {
#pragma unroll
            for (int v = 0; v < 8; ++v) {
                if (nb0 + half * 8 + v < Ntot) op0[(size_t)v * OC] = acc0[v];
                if (nb1 + half * 8 + v < Ntot) op1[(size_t)v * OC] = acc1[v];
            }
        }
    }
}

// ---------------------------------------------------------------------------
// Main kernel: 64 nodes/block, 8 waves = {gemm} x {row group} x {ct half}.
// ---------------------------------------------------------------------------
__global__ __launch_bounds__(256)
void mesh_fused_kernel(const float* __restrict__ spatial,     // [N,64]
                       const float* __restrict__ structural,  // [N,131]
                       const int*   __restrict__ nbr,         // [N,3]
                       const _Float16* __restrict__ wp,       // packed Wc|Wa (+bias)
                       float* __restrict__ out,               // [2,N,256]
                       long N) {
    __shared__ _Float16 sA[ROWS * SA1];   // concat features, f16, K-padded
    __shared__ _Float16 sV[ROWS * SV2];   // aggregated features, f16, K-padded
    __shared__ int      sNbr[ROWS * 3];

    const int  tid  = threadIdx.x;
    const long base = (long)blockIdx.x * ROWS;
    const bool full = (base + ROWS) <= N;

    // --- stage neighbor indices ---
    for (int i = tid; i < ROWS * 3; i += 256) {
        long node = base + (i / 3);
        sNbr[i] = (node < N) ? nbr[node * 3 + (i % 3)] : 0;
    }
    // --- stage concat(spatial, structural, 1.0) -> f16, zero-pad K to 224 ---
    for (int i = tid; i < ROWS * K1_PAD; i += 256) {
        int r = i / K1_PAD, c = i % K1_PAD;
        long node = base + r;
        float v = 0.0f;
        if (c == K1_REAL) v = 1.0f;                 // bias column
        else if (node < N) {
            if (c < 64)            v = __builtin_nontemporal_load(&spatial[node * 64 + c]);
            else if (c < K1_REAL)  v = structural[node * K2_REAL + (c - 64)];
        }
        sA[r * SA1 + c] = (_Float16)v;
    }
    __syncthreads();
    // --- stage vec4 = 0.25*(self + 3 gathered rows), then 1.0 -> pad to 160 ---
    for (int i = tid; i < ROWS * K2_PAD; i += 256) {
        int r = i / K2_PAD, c = i % K2_PAD;
        long node = base + r;
        float v = 0.0f;
        if (c == K2_REAL) v = 1.0f;                 // bias column
        else if (node < N && c < K2_REAL) {
            float s = structural[node * K2_REAL + c];
            long n0 = sNbr[r * 3 + 0];
            long n1 = sNbr[r * 3 + 1];
            long n2 = sNbr[r * 3 + 2];
            s += structural[n0 * K2_REAL + c];
            s += structural[n1 * K2_REAL + c];
            s += structural[n2 * K2_REAL + c];
            v = s * 0.25f;
        }
        sV[r * SV2 + c] = (_Float16)v;
    }
    __syncthreads();

    // --- compute ---
    const int wave  = tid >> 5;
    const int lane  = tid & 31;
    const int which = wave >> 2;        // 0: GEMM1, 1: GEMM2
    const int rg    = (wave >> 1) & 1;  // row group: 32 rows
    const int ch    = wave & 1;         // column-tile half: 8 ctiles
    const long nb0  = base + rg * 32;
    const long nb1  = nb0 + 16;

    if (which == 0) {
        const _Float16* r0 = &sA[(rg * 32) * SA1];
        const _Float16* r1 = &sA[(rg * 32 + 16) * SA1];
        if (full)
            wave_gemm<KC1, true >(r0, r1, SA1, wp, ch * 8, out, nb0, nb1, N, lane);
        else
            wave_gemm<KC1, false>(r0, r1, SA1, wp, ch * 8, out, nb0, nb1, N, lane);
    } else {
        float* out2 = out + (size_t)N * OC;
        const _Float16* wp2 = wp + (size_t)F1 * 512;
        const _Float16* r0 = &sV[(rg * 32) * SV2];
        const _Float16* r1 = &sV[(rg * 32 + 16) * SV2];
        if (full)
            wave_gemm<KC2, true >(r0, r1, SV2, wp2, ch * 8, out2, nb0, nb1, N, lane);
        else
            wave_gemm<KC2, false>(r0, r1, SV2, wp2, ch * 8, out2, nb0, nb1, N, lane);
    }
}

extern "C" void kernel_launch(void* const* d_in, const int* in_sizes, int n_in,
                              void* d_out, int out_size, void* d_ws, size_t ws_size,
                              hipStream_t stream) {
    const float* spatial    = (const float*)d_in[0];
    const float* structural = (const float*)d_in[1];
    const int*   neighbour  = (const int*)  d_in[2];
    const float* Wc         = (const float*)d_in[3];
    const float* bc         = (const float*)d_in[4];
    const float* Wa         = (const float*)d_in[5];
    const float* ba         = (const float*)d_in[6];
    float*       out        = (float*)d_out;
    _Float16*    wp         = (_Float16*)d_ws;   // (F1+F2)*512 halves = 192 KB

    const long N = in_sizes[0] / 64;

    // 1) pack weights+bias into f16 WMMA B-fragment layout (deterministic)
    {
        int total  = (F1 + F2) * 512;
        int blocks = (total + 255) / 256;
        mesh_pack_weights<<<blocks, 256, 0, stream>>>(Wc, bc, Wa, ba, wp);
    }
    // 2) fused gather + dual-GEMM
    {
        int blocks = (int)((N + ROWS - 1) / ROWS);
        mesh_fused_kernel<<<blocks, 256, 0, stream>>>(
            spatial, structural, neighbour, wp, out, N);
    }
}
